// DBLSTMPredictor_46823733461800
// MI455X (gfx1250) — compile-verified, matrix operand order checked by reading, not run on hardware
//
#include <hip/hip_runtime.h>

typedef __attribute__((ext_vector_type(16))) __bf16 bf16x16;
typedef __attribute__((ext_vector_type(8)))  __bf16 bf16x8;
typedef __attribute__((ext_vector_type(8)))  float  f32x8;
typedef __attribute__((ext_vector_type(4)))  float  f32x4;
typedef __attribute__((ext_vector_type(2)))  float  f32x2;

#define B_TOT 16384
#define SQ    64
#define D_IN  4
#define H     32
#define DH    16
#define CF    16
#define WAVES 2
#define LDS_PER_WAVE 32768
#define WPK_BYTES (8 * 32 * 32)   // 8 fragment-sets x 32 lanes x 32 bytes = 8KB

// CDNA5 16-bit A(16x32)/B(32x16) fragment K-map (ISA 7.12.2):
// vector elements 0..7  <-> K = hoff .. hoff+7      (hoff = (lane>=16) ? 8 : 0)
// vector elements 8..15 <-> K = 16+hoff .. 16+hoff+7
// => each fragment is two contiguous 16-byte chunks.

__device__ __forceinline__ bf16x16 join8(bf16x8 lo, bf16x8 hi) {
    bf16x16 a;
#pragma unroll
    for (int i = 0; i < 8; ++i) { a[i] = lo[i]; a[8 + i] = hi[i]; }
    return a;
}

__device__ __forceinline__ bf16x8 zero8() {
    bf16x8 z;
#pragma unroll
    for (int i = 0; i < 8; ++i) z[i] = (__bf16)0.0f;
    return z;
}

__device__ __forceinline__ f32x4 zero4f() { f32x4 z = {}; return z; }

__device__ __forceinline__ bf16x16 load_frag32(const __bf16* row, int hoff) {
    const __bf16* p = row + hoff;
    return join8(*(const bf16x8*)p, *(const bf16x8*)(p + 16));
}

__device__ __forceinline__ bf16x16 load_frag32_f32(const float* row, int hoff) {
    const float* p = row + hoff;
    f32x4 c0 = *(const f32x4*)(p);
    f32x4 c1 = *(const f32x4*)(p + 4);
    f32x4 c2 = *(const f32x4*)(p + 16);
    f32x4 c3 = *(const f32x4*)(p + 20);
    bf16x16 a;
#pragma unroll
    for (int i = 0; i < 4; ++i) {
        a[i]      = (__bf16)c0[i];
        a[4 + i]  = (__bf16)c1[i];
        a[8 + i]  = (__bf16)c2[i];
        a[12 + i] = (__bf16)c3[i];
    }
    return a;
}

// ---- Prelude: pack wq/wk/wv/wo into per-lane WMMA B-fragment layout (bf16) ----
// wpk[(mat*2+nt)*32 + lane] = 16 bf16 fragment elements, contiguous (32B).
__global__ void pack_weights_kernel(const float* __restrict__ wq, const float* __restrict__ wk,
                                    const float* __restrict__ wv, const float* __restrict__ wo,
                                    __bf16* __restrict__ wpk) {
    const int t    = threadIdx.x;           // 0..255 = (mat,nt,lane)
    const int mat  = t >> 6;
    const int nt   = (t >> 5) & 1;
    const int lane = t & 31;
    const float* W = (mat == 0) ? wq : (mat == 1) ? wk : (mat == 2) ? wv : wo;
    const int hoff = (lane >> 4) * 8;
    const int n    = nt * 16 + (lane & 15);
    __bf16* dst = wpk + (size_t)t * 16;
#pragma unroll
    for (int i = 0; i < 8; ++i) {
        dst[i]     = (__bf16)W[n * H + hoff + i];          // K = hoff..hoff+7
        dst[8 + i] = (__bf16)W[n * H + 16 + hoff + i];     // K = 16+hoff..
    }
}

__global__ __launch_bounds__(32 * WAVES) void dblstm_fused_kernel(
    const float* __restrict__ x,
    const float* __restrict__ c1w, const float* __restrict__ c1b,
    const float* __restrict__ c2w, const float* __restrict__ c2b,
    const float* __restrict__ cpw, const float* __restrict__ cpb,
    const float* __restrict__ ipw, const float* __restrict__ ipb,
    const float* __restrict__ wq,  const float* __restrict__ bq,
    const float* __restrict__ wk,  const float* __restrict__ bk,
    const float* __restrict__ wv,  const float* __restrict__ bv,
    const float* __restrict__ wo,  const float* __restrict__ bo,
    const float* __restrict__ lng, const float* __restrict__ lnb,
    const float* __restrict__ gw,  const float* __restrict__ gb,
    const float* __restrict__ pw,  const float* __restrict__ pb,
    const float* __restrict__ h1w, const float* __restrict__ h1b,
    const float* __restrict__ h2w, const float* __restrict__ h2b,
    const __bf16* __restrict__ wpk, const int use_pk,
    float* __restrict__ out)
{
    __shared__ char lds_all[WAVES * LDS_PER_WAVE];

    const int wave = threadIdx.x >> 5;
    const int lane = threadIdx.x & 31;
    const int b    = blockIdx.x * WAVES + wave;
    const int hi   = lane >> 4;
    const int ln   = lane & 15;
    const int hoff = hi * 8;

    char*    lds    = lds_all + wave * LDS_PER_WAVE;
    __bf16*  hS     = (__bf16*)(lds + 0);        // 64x32 bf16  h
    __bf16*  kS     = (__bf16*)(lds + 4096);     // 64x32 bf16  k
    __bf16*  vT     = (__bf16*)(lds + 8192);     // 32x64 bf16  v transposed [d][s]
    __bf16*  qS     = (__bf16*)(lds + 12288);    // 64x32 bf16  q, then ctx
    float*   scoreS = (float*) (lds + 16384);    // 64x64 f32
    float*   yS     = (float*) (lds + 16384);    // 64x32 f32 (reuse)
    float*   conv1S = (float*) (lds + 24576);    // 64x16 f32 [s][c]
    float*   smallS = (float*) (lds + 28672);
    float*   combL   = smallS;        // 64
    float*   pooledL = smallS + 64;   // 16
    float*   gatedL  = smallS + 80;   // 64
    float*   fusedL  = smallS + 144;  // 32
    float*   tL      = smallS + 176;  // 16

    const float* xb = x + (size_t)b * SQ * D_IN;

    // ===== Phase 1: h = x @ ipw^T + ipb  (fp32 WMMA 16x16x4) =====
#if __has_builtin(__builtin_amdgcn_wmma_f32_16x16x4_f32)
    {
#pragma unroll
        for (int nt = 0; nt < 2; ++nt) {
            const int n = nt * 16 + ln;
            f32x2 bfrag = *(const f32x2*)(ipw + n * D_IN + 2 * hi);
            const float bias = ipb[n];
#pragma unroll
            for (int mt = 0; mt < 4; ++mt) {
                f32x2 afrag = *(const f32x2*)(xb + (mt * 16 + ln) * D_IN + 2 * hi);
                f32x8 c = {};
                c = __builtin_amdgcn_wmma_f32_16x16x4_f32(false, afrag, false, bfrag,
                                                          (short)0, c, false, false);
                __bf16* dst = hS + (mt * 16 + 8 * hi) * H + n;
#pragma unroll
                for (int r = 0; r < 8; ++r) dst[r * H] = (__bf16)(c[r] + bias);
            }
        }
    }
#else
    {
        for (int idx = lane; idx < SQ * H; idx += 32) {
            int m = idx / H, n = idx % H;
            float acc = ipb[n];
            for (int k = 0; k < D_IN; ++k) acc += xb[m * D_IN + k] * ipw[n * D_IN + k];
            hS[m * H + n] = (__bf16)acc;
        }
    }
#endif
    __syncthreads();

    // ===== Phase 2: q/k/v = h @ W^T + b  (bf16 WMMA 16x16x32) =====
    {
        bf16x16 ah[4];
#pragma unroll
        for (int mt = 0; mt < 4; ++mt)
            ah[mt] = load_frag32(hS + (mt * 16 + ln) * H, hoff);

        const float* Ws[3] = {wq, wk, wv};
        const float* Bi[3] = {bq, bk, bv};
#pragma unroll
        for (int p = 0; p < 3; ++p) {
            const float* bi = Bi[p];
#pragma unroll
            for (int nt = 0; nt < 2; ++nt) {
                const int n = nt * 16 + ln;
                bf16x16 bf;
                if (use_pk) bf = ((const bf16x16*)wpk)[(p * 2 + nt) * 32 + lane];
                else        bf = load_frag32_f32(Ws[p] + n * H, hoff);
                const float bn = bi[n];
#pragma unroll
                for (int mt = 0; mt < 4; ++mt) {
                    f32x8 c = {};
                    c = __builtin_amdgcn_wmma_f32_16x16x32_bf16(false, ah[mt], false, bf,
                                                                (short)0, c, false, false);
                    if (p == 2) {
                        bf16x8 pk;
#pragma unroll
                        for (int r = 0; r < 8; ++r) pk[r] = (__bf16)(c[r] + bn);
                        *(bf16x8*)(vT + n * SQ + mt * 16 + 8 * hi) = pk; // one b128 store
                    } else {
                        __bf16* dst = (p == 0 ? qS : kS) + (mt * 16 + 8 * hi) * H + n;
#pragma unroll
                        for (int r = 0; r < 8; ++r) dst[r * H] = (__bf16)(c[r] + bn);
                    }
                }
            }
        }
    }
    __syncthreads();

    // ===== Phase 3: per-head attention =====
#pragma unroll 1
    for (int hd = 0; hd < 2; ++hd) {
        { // scores = (q_h @ k_h^T) * 0.25 ; K=16 zero-padded to 32
            bf16x8 z = zero8();
            bf16x16 aq[4];
#pragma unroll
            for (int mt = 0; mt < 4; ++mt)
                aq[mt] = join8(*(const bf16x8*)(qS + (mt * 16 + ln) * H + hd * DH + hoff), z);
#pragma unroll
            for (int nt = 0; nt < 4; ++nt) {
                const int n = nt * 16 + ln;
                bf16x16 bf = join8(*(const bf16x8*)(kS + n * H + hd * DH + hoff), z);
#pragma unroll
                for (int mt = 0; mt < 4; ++mt) {
                    f32x8 c = {};
                    c = __builtin_amdgcn_wmma_f32_16x16x32_bf16(false, aq[mt], false, bf,
                                                                (short)0, c, false, false);
                    float* dst = scoreS + (mt * 16 + 8 * hi) * SQ + n;
#pragma unroll
                    for (int r = 0; r < 8; ++r) dst[r * SQ] = c[r] * 0.25f;
                }
            }
        }

        // softmax rows in place, float4-vectorized (each lane owns 2 rows)
#pragma unroll
        for (int rr = 0; rr < 2; ++rr) {
            f32x4* sr = (f32x4*)(scoreS + (lane + rr * 32) * SQ);
            f32x4 m4 = sr[0];
#pragma unroll
            for (int j = 1; j < 16; ++j) {
                f32x4 v = sr[j];
#pragma unroll
                for (int i = 0; i < 4; ++i) m4[i] = fmaxf(m4[i], v[i]);
            }
            float mx = fmaxf(fmaxf(m4[0], m4[1]), fmaxf(m4[2], m4[3]));
            f32x4 s4 = {};
#pragma unroll
            for (int j = 0; j < 16; ++j) {
                f32x4 v = sr[j];
#pragma unroll
                for (int i = 0; i < 4; ++i) v[i] = __expf(v[i] - mx);
                sr[j] = v;
                s4 += v;
            }
            float inv = 1.0f / (s4[0] + s4[1] + s4[2] + s4[3]);
#pragma unroll
            for (int j = 0; j < 16; ++j) sr[j] *= inv;
        }

        { // ctx_h = attn @ v_h ; K=64 via two chained K=32 WMMAs
            bf16x16 bv0[2];
#pragma unroll
            for (int kc = 0; kc < 2; ++kc)
                bv0[kc] = load_frag32(vT + (hd * DH + ln) * SQ + kc * 32, hoff);
#pragma unroll
            for (int mt = 0; mt < 4; ++mt) {
                f32x8 c = {};
#pragma unroll
                for (int kc = 0; kc < 2; ++kc) {
                    bf16x16 af = load_frag32_f32(scoreS + (mt * 16 + ln) * SQ + kc * 32, hoff);
                    c = __builtin_amdgcn_wmma_f32_16x16x32_bf16(false, af, false, bv0[kc],
                                                                (short)0, c, false, false);
                }
                __bf16* dst = qS + (mt * 16 + 8 * hi) * H + hd * DH + ln;
#pragma unroll
                for (int r = 0; r < 8; ++r) dst[r * H] = (__bf16)c[r];
            }
        }
    }
    __syncthreads();

    // ===== Phase 4: attn_out = ctx @ wo^T + bo ; residual ; LayerNorm ; mean over S =====
    {
        bf16x16 actx[4];
#pragma unroll
        for (int mt = 0; mt < 4; ++mt)
            actx[mt] = load_frag32(qS + (mt * 16 + ln) * H, hoff);
#pragma unroll
        for (int nt = 0; nt < 2; ++nt) {
            const int n = nt * 16 + ln;
            bf16x16 bf;
            if (use_pk) bf = ((const bf16x16*)wpk)[(6 + nt) * 32 + lane];
            else        bf = load_frag32_f32(wo + n * H, hoff);
            const float bn = bo[n];
#pragma unroll
            for (int mt = 0; mt < 4; ++mt) {
                f32x8 c = {};
                c = __builtin_amdgcn_wmma_f32_16x16x32_bf16(false, actx[mt], false, bf,
                                                            (short)0, c, false, false);
                float*        yd  = yS + (mt * 16 + 8 * hi) * H + n;
                const __bf16* hd2 = hS + (mt * 16 + 8 * hi) * H + n;
#pragma unroll
                for (int r = 0; r < 8; ++r) yd[r * H] = c[r] + bn + (float)hd2[r * H];
            }
        }
        f32x4 lngv[8], lnbv[8];
#pragma unroll
        for (int j = 0; j < 8; ++j) {
            lngv[j] = *(const f32x4*)(lng + 4 * j);
            lnbv[j] = *(const f32x4*)(lnb + 4 * j);
        }
#pragma unroll
        for (int rr = 0; rr < 2; ++rr) {
            f32x4* yr = (f32x4*)(yS + (lane + rr * 32) * H);
            f32x4 acc = {};
#pragma unroll
            for (int j = 0; j < 8; ++j) acc += yr[j];
            float mu = (acc[0] + acc[1] + acc[2] + acc[3]) * (1.0f / H);
            f32x4 vacc = {};
#pragma unroll
            for (int j = 0; j < 8; ++j) {
                f32x4 d = yr[j];
#pragma unroll
                for (int i = 0; i < 4; ++i) { float t = d[i] - mu; vacc[i] += t * t; }
            }
            float var = (vacc[0] + vacc[1] + vacc[2] + vacc[3]) * (1.0f / H);
            float inv = rsqrtf(var + 1e-5f);
#pragma unroll
            for (int j = 0; j < 8; ++j) {
                f32x4 d = yr[j];
#pragma unroll
                for (int i = 0; i < 4; ++i)
                    d[i] = (d[i] - mu) * inv * lngv[j][i] + lnbv[j][i];
                yr[j] = d;
            }
        }
        float att_feat = 0.0f;
        for (int s = 0; s < SQ; ++s) att_feat += yS[s * H + lane];
        combL[H + lane] = att_feat * (1.0f / SQ);
    }

    // ===== Phase 5: Conv branch (rolling-window, f32x4-vectorized) =====
    {
        const int c  = lane & 15;
        const int s0 = hi * 32;
        f32x4 w1v[3];
#pragma unroll
        for (int t = 0; t < 3; ++t)
#pragma unroll
            for (int i = 0; i < 4; ++i) w1v[t][i] = c1w[(c * D_IN + i) * 3 + t];
        const float b1 = c1b[c];
        f32x4 xm = (s0 == 0) ? zero4f() : *(const f32x4*)(xb + (s0 - 1) * D_IN);
        f32x4 x0 = *(const f32x4*)(xb + s0 * D_IN);
        for (int s = s0; s < s0 + 32; ++s) {
            f32x4 xp = (s + 1 < SQ) ? *(const f32x4*)(xb + (s + 1) * D_IN) : zero4f();
            float acc = b1;
#pragma unroll
            for (int i = 0; i < 4; ++i)
                acc += w1v[0][i] * xm[i] + w1v[1][i] * x0[i] + w1v[2][i] * xp[i];
            conv1S[s * CF + c] = fmaxf(acc, 0.0f);   // [s][c] layout
            xm = x0; x0 = xp;
        }
        f32x4 w2v[3][4];
#pragma unroll
        for (int t = 0; t < 3; ++t)
#pragma unroll
            for (int j = 0; j < 4; ++j)
#pragma unroll
                for (int i = 0; i < 4; ++i)
                    w2v[t][j][i] = c2w[(c * CF + 4 * j + i) * 3 + t];
        const float b2 = c2b[c];
        f32x4 rm[4], r0[4], rp[4];
#pragma unroll
        for (int j = 0; j < 4; ++j) {
            rm[j] = (s0 == 0) ? zero4f() : ((const f32x4*)(conv1S + (s0 - 1) * CF))[j];
            r0[j] = ((const f32x4*)(conv1S + s0 * CF))[j];
        }
        float pooled = 0.0f;
        for (int s = s0; s < s0 + 32; ++s) {
            float acc = b2;
#pragma unroll
            for (int j = 0; j < 4; ++j) {
                rp[j] = (s + 1 < SQ) ? ((const f32x4*)(conv1S + (s + 1) * CF))[j] : zero4f();
#pragma unroll
                for (int i = 0; i < 4; ++i)
                    acc += w2v[0][j][i] * rm[j][i] + w2v[1][j][i] * r0[j][i]
                         + w2v[2][j][i] * rp[j][i];
            }
            pooled += fmaxf(acc, 0.0f);
#pragma unroll
            for (int j = 0; j < 4; ++j) { rm[j] = r0[j]; r0[j] = rp[j]; }
        }
        pooled += __shfl_xor(pooled, 16, 32);
        if (hi == 0) pooledL[c] = pooled * (1.0f / SQ);
        float cf = cpb[lane];
        const f32x4* cw = (const f32x4*)(cpw + lane * CF);
#pragma unroll
        for (int j = 0; j < 4; ++j) {
            f32x4 wv = cw[j];
            f32x4 pv = ((const f32x4*)pooledL)[j];
#pragma unroll
            for (int i = 0; i < 4; ++i) cf += wv[i] * pv[i];
        }
        combL[lane] = cf;
    }
    __syncthreads();

    // ===== Phase 6: DBMM gate + projection + output head =====
    {
        f32x4 combv[16];
#pragma unroll
        for (int j = 0; j < 16; ++j) combv[j] = ((const f32x4*)combL)[j];
#pragma unroll
        for (int rr = 0; rr < 2; ++rr) {
            const int i = lane + rr * 32;
            float g = gb[i];
            const f32x4* gr = (const f32x4*)(gw + i * 2 * H);
#pragma unroll
            for (int j = 0; j < 16; ++j) {
                f32x4 wv = gr[j];
#pragma unroll
                for (int q = 0; q < 4; ++q) g += wv[q] * combv[j][q];
            }
            gatedL[i] = combL[i] * (1.0f / (1.0f + __expf(-g)));
        }
        float f = pb[lane];
        const f32x4* pr = (const f32x4*)(pw + lane * 2 * H);
#pragma unroll
        for (int j = 0; j < 16; ++j) {
            f32x4 wv = pr[j];
            f32x4 gv = ((const f32x4*)gatedL)[j];
#pragma unroll
            for (int q = 0; q < 4; ++q) f += wv[q] * gv[q];
        }
        fusedL[lane] = f;
        if (lane < 16) {
            float t = h1b[lane];
            const f32x4* hr = (const f32x4*)(h1w + lane * H);
#pragma unroll
            for (int j = 0; j < 8; ++j) {
                f32x4 wv = hr[j];
                f32x4 fv = ((const f32x4*)fusedL)[j];
#pragma unroll
                for (int q = 0; q < 4; ++q) t += wv[q] * fv[q];
            }
            tL[lane] = fmaxf(t, 0.0f);
        }
        if (lane < 2) {
            float o = h2b[lane];
            for (int j = 0; j < 16; ++j) o += h2w[lane * 16 + j] * tL[j];
            out[(size_t)b * 2 + lane] = o;
        }
    }
}

extern "C" void kernel_launch(void* const* d_in, const int* in_sizes, int n_in,
                              void* d_out, int out_size, void* d_ws, size_t ws_size,
                              hipStream_t stream) {
    (void)in_sizes; (void)n_in; (void)out_size;
    const float* p[27];
    for (int i = 0; i < 27; ++i) p[i] = (const float*)d_in[i];
    dim3 grid(B_TOT / WAVES), block(32 * WAVES);
    const int use_pk = (ws_size >= WPK_BYTES) ? 1 : 0;
    const __bf16* wpk = use_pk ? (const __bf16*)d_ws : (const __bf16*)nullptr;
    if (use_pk) {
        hipLaunchKernelGGL(pack_weights_kernel, dim3(1), dim3(256), 0, stream,
                           p[9], p[11], p[13], p[15], (__bf16*)d_ws);
    }
    hipLaunchKernelGGL(dblstm_fused_kernel, grid, block, 0, stream,
        p[0],  p[1],  p[2],  p[3],  p[4],  p[5],  p[6],  p[7],  p[8],
        p[9],  p[10], p[11], p[12], p[13], p[14], p[15], p[16], p[17],
        p[18], p[19], p[20], p[21], p[22], p[23], p[24], p[25], p[26],
        wpk, use_pk, (float*)d_out);
}
